// SparseCodingLayer_First_23270132809930
// MI455X (gfx1250) — matrix operand order repaired and checked by reading, not compiled, not run on hardware
//
#include <hip/hip_runtime.h>
#include <hip/hip_bf16.h>

// ---------------------------------------------------------------------------
// Fused SparseCodingLayer for MI455X (gfx1250, wave32, WMMA).
//   out  = conv1x1(topk16(conv3x3(x,We)+be), Wr)+br        [16,64,222,222]
//   loss = mean((im2col(x) - sigmoid(conv1x1(topk_h,Wd)+bd))^2)
// Everything fused per 16-pixel strip; h never touches HBM.
// ---------------------------------------------------------------------------

typedef __attribute__((ext_vector_type(16))) _Float16 v16h;
typedef __attribute__((ext_vector_type(8)))  float    v8f;

union AV { v16h v; _Float16 e[16]; unsigned u[8]; };
union CV { v8f  v; float e[8]; };
union H2 { unsigned u; _Float16 h[2]; unsigned short us[2]; };

#define B_      16
#define CIN_    3
#define HH_     224
#define WW_     224
#define F_      128
#define COUT_   64
#define HP_     222
#define WP_     222
#define KTOP_   16
#define TILES_PER_ROW_ 14            // ceil(222/16)
#define NROWS_  (B_ * HP_)           // 3552
#define NTILES_ (NROWS_ * TILES_PER_ROW_)  // 49728
#define WAVES_  4

// ws layout (dwords): [0,2048) WeT  [2048,6144) WrT  [6144,8192) WdT
//                     [8192, 8192+NTILES_) float loss partials
#define WS_WE_   0
#define WS_WR_   2048
#define WS_WD_   6144
#define WS_PART_ 8192

__device__ __forceinline__ int kpairA(int v, int hi) {
    // 16-bit A operand 16x32: v0-3 -> K0..7, v4-7 -> K16..23 (lo half); +8 for hi half
    return (v < 4 ? 2 * v : 16 + 2 * (v - 4)) + 8 * hi;
}

__device__ __forceinline__ unsigned pack2(float a, float b) {
    H2 p; p.h[0] = (_Float16)a; p.h[1] = (_Float16)b; return p.u;
}

// --- Kernel 1: repack fp32 weights into wave32 WMMA B-operand f16 layout ----
// B operand 32x16 (KxN): lane n (0..15) holds K=0..7,16..23 (2 per VGPR), lane n+16
// holds K=8..15,24..31.  Entry index e = v*32+lane, per tile of 256 dwords.
__global__ __launch_bounds__(256) void pack_weights(const float* __restrict__ We,
                                                    const float* __restrict__ Wr,
                                                    const float* __restrict__ Wd,
                                                    unsigned* __restrict__ ws) {
    int idx = blockIdx.x * 256 + threadIdx.x;   // 0 .. 8191
    int e    = idx & 255;
    int v    = e >> 5;
    int lane = e & 31;
    int n    = lane & 15;
    int k0   = (v < 4 ? 2 * v : 16 + 2 * (v - 4)) + ((lane >> 4) ? 8 : 0);

    if (idx < WS_WR_) {                        // WeT: 8 f-tiles, K=27 padded to 32
        int t = idx >> 8;
        int f = t * 16 + n;
        float lo = (k0     < 27) ? We[f * 27 + k0]     : 0.f;
        float hi = (k0 + 1 < 27) ? We[f * 27 + k0 + 1] : 0.f;
        ws[WS_WE_ + idx] = pack2(lo, hi);
    } else if (idx < WS_WD_) {                 // WrT: 4 n-tiles x 4 k-chunks
        int q  = idx - WS_WR_;
        int nt = q >> 10;
        int kc = (q >> 8) & 3;
        int cout = nt * 16 + n;
        int k    = kc * 32 + k0;
        ws[idx] = pack2(Wr[cout * F_ + k], Wr[cout * F_ + k + 1]);
    } else {                                   // WdT: 2 n-tiles (27 pad 32) x 4 k-chunks
        int q  = idx - WS_WD_;
        int nt = q >> 10;
        int kc = (q >> 8) & 3;
        int nr = nt * 16 + n;
        int k  = kc * 32 + k0;
        float lo = (nr < 27) ? Wd[nr * F_ + k]     : 0.f;
        float hi = (nr < 27) ? Wd[nr * F_ + k + 1] : 0.f;
        ws[idx] = pack2(lo, hi);
    }
}

// --- Kernel 2: fused conv3x3 -> topk -> {conv1x1 main, aux head + MSE} ------
__global__ __launch_bounds__(WAVES_ * 32) void fused_sparse_coding(
        const float* __restrict__ x, const float* __restrict__ be,
        const float* __restrict__ br, const float* __restrict__ bd,
        const unsigned* __restrict__ ws, float* __restrict__ out,
        float* __restrict__ partials) {
    __shared__ float          s_h [WAVES_][16 * F_];   // conv1 result, fp32
    __shared__ unsigned short s_hf[WAVES_][16 * F_];   // masked h, f16

    const int lane = threadIdx.x & 31;
    const int wave = threadIdx.x >> 5;
    const int tile = blockIdx.x * WAVES_ + wave;

    const int row = tile / TILES_PER_ROW_;
    const int tc  = tile % TILES_PER_ROW_;
    const int b   = row / HP_;
    const int oh  = row % HP_;
    const int ow0 = tc * 16;

    const int m  = lane & 15;
    const int hi = lane >> 4;
    const int owm = (ow0 + m > WP_ - 1) ? (WP_ - 1) : (ow0 + m);  // clamp pad pixels

    // ---- build conv1 A operand (im2col, K=27 padded to 32, f16) ----
    AV A1;
#pragma unroll
    for (int v = 0; v < 8; ++v) {
        int kb = kpairA(v, hi);
#pragma unroll
        for (int s = 0; s < 2; ++s) {
            int k = kb + s;
            float val = 0.f;
            if (k < 27) {
                int c = k / 9, rr = k % 9, dh = rr / 3, dw = rr % 3;
                val = x[((size_t)(b * CIN_ + c) * HH_ + (oh + dh)) * WW_ + (owm + dw)];
            }
            A1.e[2 * v + s] = (_Float16)val;
        }
    }

    // ---- conv1: 8 WMMAs over the 128 feature channels ----
#pragma unroll
    for (int t = 0; t < 8; ++t) {
        CV acc;
        float bias = be[t * 16 + m];
#pragma unroll
        for (int r = 0; r < 8; ++r) acc.e[r] = bias;
        AV Bw;
#pragma unroll
        for (int v = 0; v < 8; ++v) Bw.u[v] = ws[WS_WE_ + (t * 8 + v) * 32 + lane];
        acc.v = __builtin_amdgcn_wmma_f32_16x16x32_f16(false, A1.v, false, Bw.v,
                                                       (short)0, acc.v, false, false);
#pragma unroll
        for (int r = 0; r < 8; ++r)
            s_h[wave][(r + 8 * hi) * F_ + t * 16 + m] = acc.e[r];
    }
    __builtin_amdgcn_wave_barrier();   // DS in-order within wave; fence scheduling

    // ---- exact top-16 of 128 by (|v|, -idx), matching lax.top_k ----
    if (lane < 16) {
        const float* rowp = &s_h[wave][lane * F_];
        float curV = 3.4e38f; int curI = -1;
#pragma unroll 1
        for (int s = 0; s < KTOP_; ++s) {
            float best = -1.f; int bestI = 0;
#pragma unroll 4
            for (int f = 0; f < F_; ++f) {
                float a = fabsf(rowp[f]);
                bool elig = (a < curV) || (a == curV && f > curI);
                if (elig && a > best) { best = a; bestI = f; }
            }
            curV = best; curI = bestI;
        }
#pragma unroll 4
        for (int f = 0; f < F_; ++f) {
            float vv = rowp[f];
            float a  = fabsf(vv);
            bool keep = (a > curV) || (a == curV && f <= curI);
            H2 p; p.h[0] = keep ? (_Float16)vv : (_Float16)0.f;
            s_hf[wave][lane * F_ + f] = p.us[0];
        }
    }
    __builtin_amdgcn_wave_barrier();

    // ---- rebuild masked h as 4 A K-chunks (f16) ----
    AV A2[4];
#pragma unroll
    for (int c = 0; c < 4; ++c)
#pragma unroll
        for (int v = 0; v < 8; ++v) {
            int kb = c * 32 + kpairA(v, hi);           // even -> 4B aligned
            A2[c].u[v] = *(const unsigned*)&s_hf[wave][m * F_ + kb];
        }

    // ---- main branch: 1x1 conv to 64 channels (16 WMMAs) ----
#pragma unroll
    for (int j = 0; j < 4; ++j) {
        CV d;
        float bb = br[j * 16 + m];
#pragma unroll
        for (int r = 0; r < 8; ++r) d.e[r] = bb;
#pragma unroll
        for (int c = 0; c < 4; ++c) {
            AV Bw;
#pragma unroll
            for (int v = 0; v < 8; ++v)
                Bw.u[v] = ws[WS_WR_ + ((j * 4 + c) * 8 + v) * 32 + lane];
            d.v = __builtin_amdgcn_wmma_f32_16x16x32_f16(false, A2[c].v, false, Bw.v,
                                                         (short)0, d.v, false, false);
        }
        int cout = j * 16 + m;
        size_t obase = ((size_t)(b * COUT_ + cout) * HP_ + oh) * (size_t)WP_;
#pragma unroll
        for (int r = 0; r < 8; ++r) {
            int ow = ow0 + r + 8 * hi;
            if (ow < WP_) out[obase + ow] = d.e[r];    // 32B contiguous per lane
        }
    }

    // ---- aux head: 27 outputs (pad 32), sigmoid + MSE vs im2col(x) ----
    float lsum = 0.f;
#pragma unroll
    for (int j = 0; j < 2; ++j) {
        int n = j * 16 + m;
        CV d;
        float bb = (n < 27) ? bd[n] : 0.f;
#pragma unroll
        for (int r = 0; r < 8; ++r) d.e[r] = bb;
#pragma unroll
        for (int c = 0; c < 4; ++c) {
            AV Bw;
#pragma unroll
            for (int v = 0; v < 8; ++v)
                Bw.u[v] = ws[WS_WD_ + ((j * 4 + c) * 8 + v) * 32 + lane];
            d.v = __builtin_amdgcn_wmma_f32_16x16x32_f16(false, A2[c].v, false, Bw.v,
                                                         (short)0, d.v, false, false);
        }
        if (n < 27) {
            int cc = n / 9, rr = n % 9, dh = rr / 3, dw = rr % 3;
            const float* xp = &x[((size_t)(b * CIN_ + cc) * HH_ + (oh + dh)) * WW_];
#pragma unroll
            for (int r = 0; r < 8; ++r) {
                int ow = ow0 + r + 8 * hi;
                if (ow < WP_) {
                    float a   = 1.f / (1.f + __expf(-d.e[r]));
                    float tgt = xp[ow + dw];
                    float df  = tgt - a;
                    lsum += df * df;
                }
            }
        }
    }
    // wave reduction (wave32) -> one deterministic partial per tile
#pragma unroll
    for (int off = 16; off > 0; off >>= 1) lsum += __shfl_down(lsum, off, 32);
    if (lane == 0) partials[tile] = lsum;
}

// --- Kernel 3: fixed-order reduction of loss partials (deterministic) -------
__global__ __launch_bounds__(512) void reduce_loss(const float* __restrict__ partials,
                                                   float* __restrict__ out_loss,
                                                   int n, float scale) {
    __shared__ float sm[512];
    int tid = threadIdx.x;
    float s = 0.f;
    for (int i = tid; i < n; i += 512) s += partials[i];
    sm[tid] = s;
    __syncthreads();
    for (int o = 256; o > 0; o >>= 1) {
        if (tid < o) sm[tid] += sm[tid + o];
        __syncthreads();
    }
    if (tid == 0) out_loss[0] = sm[0] * scale;
}

extern "C" void kernel_launch(void* const* d_in, const int* in_sizes, int n_in,
                              void* d_out, int out_size, void* d_ws, size_t ws_size,
                              hipStream_t stream) {
    const float* x  = (const float*)d_in[0];
    const float* We = (const float*)d_in[1];
    const float* be = (const float*)d_in[2];
    const float* Wd = (const float*)d_in[3];
    const float* bd = (const float*)d_in[4];
    const float* Wr = (const float*)d_in[5];
    const float* br = (const float*)d_in[6];
    (void)in_sizes; (void)n_in; (void)ws_size;   // k is fixed at 16 in the reference

    float* out = (float*)d_out;
    unsigned* wsu   = (unsigned*)d_ws;
    float* partials = (float*)d_ws + WS_PART_;

    pack_weights<<<32, 256, 0, stream>>>(We, Wr, Wd, wsu);

    fused_sparse_coding<<<NTILES_ / WAVES_, WAVES_ * 32, 0, stream>>>(
        x, be, br, bd, wsu, out, partials);

    const float scale = 1.0f / (float)((long long)B_ * 27 * HP_ * WP_);
    reduce_loss<<<1, 512, 0, stream>>>(partials, out + (out_size - 1), NTILES_, scale);
}
